// CF_15358803050593
// MI455X (gfx1250) — matrix operand order repaired and checked by reading, not compiled, not run on hardware
//
#include <hip/hip_runtime.h>
#include <stdint.h>

#define DIMS 32
#define S_MC 2
#define N_USERS 300000
#define N_ITEMS 200000
#define NM (N_USERS + N_ITEMS)
#define B_SAMP 524288
#define NBLK ((NM + 255) / 256)

typedef __attribute__((ext_vector_type(16))) _Float16 v16h;
typedef __attribute__((ext_vector_type(8)))  float    v8f;

// ---------------- deterministic counter-based Gaussian sampler ----------------
__device__ __forceinline__ uint32_t mix32(uint32_t h) {
    h ^= h >> 16; h *= 0x7feb352dU;
    h ^= h >> 15; h *= 0x846ca68bU;
    h ^= h >> 16;
    return h;
}

// Returns the two Box-Muller outputs: .x used for MC sample s=0, .y for s=1.
__device__ __forceinline__ float2 nrm_pair(uint32_t e, uint32_t d, uint32_t strm) {
    uint32_t h1 = mix32(e * 0x9E3779B1u + d * 0x85EBCA77u + strm * 0xC2B2AE3Du + 0x27220A95u);
    uint32_t h2 = mix32(h1 + 0x9E3779B9u);
    float u1 = fmaxf((float)(h1 >> 8) * (1.0f / 16777216.0f), 5.96046448e-8f);
    float u2 = (float)(h2 >> 8) * (1.0f / 16777216.0f);
    float r = __fsqrt_rn(-2.0f * __logf(u1));
    float sv, cv;
    __sincosf(6.28318530718f * u2, &sv, &cv);
    return make_float2(r * cv, r * sv);
}

__device__ __forceinline__ float scrub(float v) {   // nan_to_num(nan=1e-6)
    return (v == v) ? v : 1e-6f;
}

__device__ __forceinline__ float block_reduce256(float v) {
    __shared__ float sm[256];
    __syncthreads();
    sm[threadIdx.x] = v;
    __syncthreads();
    #pragma unroll
    for (int s = 128; s > 0; s >>= 1) {
        if ((int)threadIdx.x < s) sm[threadIdx.x] += sm[threadIdx.x + s];
        __syncthreads();
    }
    return sm[0];
}

// ---------------- counts: histogram of entity occurrences ----------------
__global__ __launch_bounds__(256) void count_kernel(const long long* __restrict__ x,
                                                    float* __restrict__ counts) {
    int t = blockIdx.x * 256 + threadIdx.x;
    if (t < 2 * B_SAMP) {
        atomicAdd(&counts[(int)x[t]], 1.0f);   // adds of 1.0 are exact -> deterministic
    }
}

// ---------------- normalizers: sum(cnt/cnt) over users / items ----------------
__global__ __launch_bounds__(256) void norm_partial_kernel(const float* __restrict__ counts,
                                                           float* __restrict__ pU,
                                                           float* __restrict__ pI) {
    int e = blockIdx.x * 256 + threadIdx.x;
    float ru = 0.0f, ri = 0.0f;
    if (e < NM) {
        float c = counts[e];
        float r = c / c;              // matches reference: user_cnt/counts (identical arrays)
        if (e < N_USERS) ru = r; else ri = r;
    }
    float su = block_reduce256(ru);
    float si = block_reduce256(ri);
    if (threadIdx.x == 0) { pU[blockIdx.x] = su; pI[blockIdx.x] = si; }
}

__global__ __launch_bounds__(256) void norm_final_kernel(const float* __restrict__ pU,
                                                         const float* __restrict__ pI,
                                                         float* __restrict__ norms) {
    float su = 0.0f, si = 0.0f;
    for (int i = threadIdx.x; i < NBLK; i += 256) { su += pU[i]; si += pI[i]; }
    su = block_reduce256(su);
    si = block_reduce256(si);
    if (threadIdx.x == 0) { norms[0] = su; norms[1] = si; }
}

// ---------------- KL: async-stream 128 MB through LDS, fixed-order two-stage reduce ----------
// Each wave stages its 32 entity rows via GLOBAL_LOAD_ASYNC_TO_LDS_B128 (ASYNCcnt-tracked,
// no VGPR staging), in two half-row phases. Rows padded to 144 B in LDS (16 B pad).
#define KL_ROWPAD 144                 // 128 B half-row + 16 B pad (16B-aligned, few conflicts)
#define KL_WAVEBUF (32 * KL_ROWPAD)   // 4608 B per wave

__global__ __launch_bounds__(256) void kl_partial_kernel(const float* __restrict__ bias_w,
                                                         const float* __restrict__ ent_w,
                                                         const float* __restrict__ counts,
                                                         const float* __restrict__ norms,
                                                         float* __restrict__ pKL,
                                                         int use_counts) {
    __shared__ __align__(16) char lbuf[8 * KL_WAVEBUF];   // 36864 B static LDS
    const int t = threadIdx.x;
    const int w = t >> 5;
    const int l = t & 31;
    const int e = blockIdx.x * 256 + t;
    const int waveBaseEnt = blockIdx.x * 256 + w * 32;    // NM % 32 == 0 -> wave-uniform range
    char* wbuf = lbuf + w * KL_WAVEBUF;

    // ---- phase 0: async-stage mu half-rows (32 entities x 128 B) ----
    #pragma unroll
    for (int i = 0; i < 8; ++i) {
        int c      = i * 32 + l;          // chunk id within wave tile
        int entIn  = c >> 3;              // 8 x 16B chunks per half-row
        int within = c & 7;
        long long srcEnt = (long long)waveBaseEnt + entIn;
        if (srcEnt >= NM) srcEnt = NM - 1;            // clamp: no OOB global reads
        unsigned long long ga = (unsigned long long)(uintptr_t)ent_w
                              + (unsigned long long)srcEnt * 256ull
                              + (unsigned)(within * 16);
        unsigned ldsoff = (unsigned)(uintptr_t)(wbuf + entIn * KL_ROWPAD + within * 16);
        asm volatile("global_load_async_to_lds_b128 %0, %1, off"
                     :: "v"(ldsoff), "v"(ga) : "memory");
    }
    asm volatile("s_wait_asynccnt 0x0" ::: "memory");

    float kle = -0.5f * (float)DIMS;
    {
        const float4* row = (const float4*)(wbuf + l * KL_ROWPAD);
        #pragma unroll
        for (int q = 0; q < 8; ++q) {                 // mu part: dims 0..31
            float4 v = row[q];
            float a = scrub(v.x), b = scrub(v.y), c = scrub(v.z), d = scrub(v.w);
            kle += 0.5f * (a * a + b * b + c * c + d * d);
        }
    }

    // DS reads of phase 0 must land before async writes of phase 1 can overwrite the rows
    asm volatile("s_wait_dscnt 0x0" ::: "memory");

    // ---- phase 1: async-stage sigma half-rows ----
    #pragma unroll
    for (int i = 0; i < 8; ++i) {
        int c      = i * 32 + l;
        int entIn  = c >> 3;
        int within = c & 7;
        long long srcEnt = (long long)waveBaseEnt + entIn;
        if (srcEnt >= NM) srcEnt = NM - 1;
        unsigned long long ga = (unsigned long long)(uintptr_t)ent_w
                              + (unsigned long long)srcEnt * 256ull
                              + (unsigned)(128 + within * 16);
        unsigned ldsoff = (unsigned)(uintptr_t)(wbuf + entIn * KL_ROWPAD + within * 16);
        asm volatile("global_load_async_to_lds_b128 %0, %1, off"
                     :: "v"(ldsoff), "v"(ga) : "memory");
    }
    asm volatile("s_wait_asynccnt 0x0" ::: "memory");

    {
        const float4* row = (const float4*)(wbuf + l * KL_ROWPAD);
        #pragma unroll
        for (int q = 0; q < 8; ++q) {                 // sigma part: dims 0..31
            float4 v = row[q];
            float a = fabsf(scrub(v.x)), b = fabsf(scrub(v.y));
            float c = fabsf(scrub(v.z)), d = fabsf(scrub(v.w));
            kle += 0.5f * (a * a + b * b + c * c + d * d);
            kle -= __logf(a) + __logf(b) + __logf(c) + __logf(d);
        }
    }

    float kl = 0.0f;
    if (e < NM) {
        float bmu = bias_w[2 * e];
        float bsc = fabsf(bias_w[2 * e + 1]);
        float klb = -__logf(bsc) + 0.5f * (bsc * bsc + bmu * bmu) - 0.5f;

        float wgt;
        if (use_counts) {
            float c = counts[e];
            float ratio = c / c;
            float fu = (e <= N_USERS) ? ((float)N_USERS / norms[0]) : 0.0f;  // ref uses <=
            float fi = (e >  N_USERS) ? ((float)N_ITEMS / norms[1]) : 0.0f;
            wgt = ratio * (fu + fi);
        } else {
            wgt = 1.0f;                 // analytic value when every count > 0
        }
        kl = (klb + kle) * wgt;
    }
    float s = block_reduce256(kl);
    if (threadIdx.x == 0) pKL[blockIdx.x] = s;
}

__global__ __launch_bounds__(256) void final_kernel(const float* __restrict__ pKL,
                                                    const float* __restrict__ alpha,
                                                    const float* __restrict__ gbm,
                                                    const float* __restrict__ gbs,
                                                    float* __restrict__ out) {
    float s = 0.0f;
    for (int i = threadIdx.x; i < NBLK; i += 256) s += pKL[i];
    s = block_reduce256(s);
    if (threadIdx.x == 0) {
        float ga  = fabsf(gbs[0]);
        float klg = -__logf(ga) + 0.5f * (ga * ga + gbm[0] * gbm[0]) - 0.5f;
        out[S_MC * B_SAMP]     = __fsqrt_rn(1.0f / fabsf(alpha[0]));   // std_dev
        out[S_MC * B_SAMP + 1] = klg + s;                              // kl_total
    }
}

// ---------------- prediction: 16 samples/wave32, dot products on the matrix pipe ----------------
// A tile: A = user embeddings (16x32 f16, row m = sample m), B = item embeddings (32x16 f16,
// col n = sample n). One v_wmma_f32_16x16x32_f16 per MC draw, accumulated; the 16 wanted dot
// products are the diagonal of C.
__global__ __launch_bounds__(256) void pred_kernel(const float* __restrict__ gbm,
                                                   const float* __restrict__ gbs,
                                                   const float* __restrict__ bias_w,
                                                   const float* __restrict__ ent_w,
                                                   const long long* __restrict__ x,
                                                   float* __restrict__ out) {
    const int wave = (int)((blockIdx.x * blockDim.x + threadIdx.x) >> 5);  // tile id
    const int lane = threadIdx.x & 31;
    const int m    = lane & 15;            // sample within tile
    const int hi   = lane >> 4;            // K-half group for WMMA layout
    const int b    = wave * 16 + m;        // B_SAMP divisible by 16

    const int eu = (int)x[2 * b + 0];
    const int ei = (int)x[2 * b + 1];

    // ---- bias part (mean over the two MC draws, via Box-Muller pair) ----
    float bmu_u = bias_w[2 * eu], bsc_u = fabsf(bias_w[2 * eu + 1]);
    float bmu_i = bias_w[2 * ei], bsc_i = fabsf(bias_w[2 * ei + 1]);
    float2 nu = nrm_pair((uint32_t)eu, 0u, 1u);
    float2 ni = nrm_pair((uint32_t)ei, 0u, 1u);
    float sumb = bmu_u + bmu_i + 0.5f * (bsc_u * (nu.x + nu.y) + bsc_i * (ni.x + ni.y));

    // ---- A matrix (user embeddings), WMMA 16-bit A layout ----
    // lane<16: K in {0..7, 16..23}; lane>=16: K in {8..15, 24..31}
    const int c0 = hi * 8;
    const int c1 = 16 + hi * 8;
    const float* up = ent_w + (size_t)eu * (2 * DIMS);
    float amu[16], asc[16];
    ((float4*)amu)[0] = *(const float4*)(up + c0);
    ((float4*)amu)[1] = *(const float4*)(up + c0 + 4);
    ((float4*)amu)[2] = *(const float4*)(up + c1);
    ((float4*)amu)[3] = *(const float4*)(up + c1 + 4);
    ((float4*)asc)[0] = *(const float4*)(up + DIMS + c0);
    ((float4*)asc)[1] = *(const float4*)(up + DIMS + c0 + 4);
    ((float4*)asc)[2] = *(const float4*)(up + DIMS + c1);
    ((float4*)asc)[3] = *(const float4*)(up + DIMS + c1 + 4);

    v16h A0, A1;
    #pragma unroll
    for (int j = 0; j < 16; ++j) {
        int dj = (j < 8) ? (c0 + j) : (c1 + j - 8);
        float mu = scrub(amu[j]);
        float sc = fabsf(scrub(asc[j]));
        float2 n = nrm_pair((uint32_t)eu, (uint32_t)dj, 2u);
        A0[j] = (_Float16)(mu + sc * n.x);
        A1[j] = (_Float16)(mu + sc * n.y);
    }

    // ---- B matrix (item embeddings), WMMA 16-bit B layout ----
    // lane<16: K in {0..7, 16..23}; lane>=16: K in {8..15, 24..31} (column n = lane&15)
    const int k0 = hi * 16;
    const float* ip = ent_w + (size_t)ei * (2 * DIMS);
    float imu[16], isc[16];
    #pragma unroll
    for (int q = 0; q < 4; ++q) {
        int kq = (q < 2) ? (c0 + 4 * q) : (c1 + 4 * (q - 2));
        ((float4*)imu)[q] = *(const float4*)(ip + kq);
        ((float4*)isc)[q] = *(const float4*)(ip + DIMS + kq);
    }

    v16h B0, B1;
    #pragma unroll
    for (int j = 0; j < 16; ++j) {
        int dj = (j < 8) ? (c0 + j) : (c1 + j - 8);
        float mu = scrub(imu[j]);
        float sc = fabsf(scrub(isc[j]));
        float2 n = nrm_pair((uint32_t)ei, (uint32_t)dj, 2u);
        B0[j] = (_Float16)(mu + sc * n.x);
        B1[j] = (_Float16)(mu + sc * n.y);
    }

    // ---- two accumulated WMMAs: C = A0*B0 + A1*B1 ----
    v8f acc = {};
    acc = __builtin_amdgcn_wmma_f32_16x16x32_f16(false, A0, false, B0, (short)0, acc, false, false);
    acc = __builtin_amdgcn_wmma_f32_16x16x32_f16(false, A1, false, B1, (short)0, acc, false, false);

    // ---- diagonal extraction: diag(m) lives in VGPR m&7 on lane (m<8 ? m : m+16) ----
    int sel = lane & 7;
    float t0 = (sel & 1) ? acc[1] : acc[0];
    float t1 = (sel & 1) ? acc[3] : acc[2];
    float t2 = (sel & 1) ? acc[5] : acc[4];
    float t3 = (sel & 1) ? acc[7] : acc[6];
    float u0 = (sel & 2) ? t1 : t0;
    float u1 = (sel & 2) ? t3 : t2;
    float dv = (sel & 4) ? u1 : u0;
    int src = (m & 8) ? (m + 16) : m;
    float dotv = __shfl(dv, src, 32);          // ds_bpermute

    // ---- global bias draws and store (lanes 0..15 own the 16 samples) ----
    float2 g  = nrm_pair(0u, 0u, 0u);
    float ga  = fabsf(gbs[0]);
    float gm  = gbm[0];
    float base = sumb + 0.5f * dotv;           // mean over S=2 draws
    if (hi == 0) {
        out[b]          = gm + ga * g.x + base;
        out[B_SAMP + b] = gm + ga * g.y + base;
    }
}

extern "C" void kernel_launch(void* const* d_in, const int* in_sizes, int n_in,
                              void* d_out, int out_size, void* d_ws, size_t ws_size,
                              hipStream_t stream) {
    const float*     alpha  = (const float*)d_in[0];
    const float*     gbm    = (const float*)d_in[1];
    const float*     gbs    = (const float*)d_in[2];
    const float*     bias_w = (const float*)d_in[3];
    const float*     ent_w  = (const float*)d_in[4];
    const long long* x      = (const long long*)d_in[5];
    float* out = (float*)d_out;
    float* ws  = (float*)d_ws;

    size_t need_full = (size_t)(NM + 3 * NBLK + 2) * sizeof(float);
    int use_counts = (ws_size >= need_full) ? 1 : 0;

    float *counts, *pU, *pI, *norms, *pKL;
    if (use_counts) {
        counts = ws;
        pU     = counts + NM;
        pI     = pU + NBLK;
        norms  = pI + NBLK;
        pKL    = norms + 2;
        hipMemsetAsync(counts, 0, (size_t)NM * sizeof(float), stream);
        count_kernel<<<(2 * B_SAMP + 255) / 256, 256, 0, stream>>>(x, counts);
        norm_partial_kernel<<<NBLK, 256, 0, stream>>>(counts, pU, pI);
        norm_final_kernel<<<1, 256, 0, stream>>>(pU, pI, norms);
    } else {
        counts = ws; norms = ws; pKL = ws;     // fallback: weight == 1 analytically
    }

    kl_partial_kernel<<<NBLK, 256, 0, stream>>>(bias_w, ent_w, counts, norms, pKL, use_counts);
    final_kernel<<<1, 256, 0, stream>>>(pKL, alpha, gbm, gbs, out);

    // 32768 tiles of 16 samples, 8 waves (tiles) per 256-thread block
    pred_kernel<<<(B_SAMP / 16) / 8, 256, 0, stream>>>(gbm, gbs, bias_w, ent_w, x, out);
}
